// _FFTConvTransposeForward_81681688035754
// MI455X (gfx1250) — matrix operand order repaired and checked by reading, not compiled, not run on hardware
//
#include <hip/hip_runtime.h>

typedef float  v2f   __attribute__((ext_vector_type(2)));
typedef float  v8f   __attribute__((ext_vector_type(8)));
typedef __bf16 v16bf __attribute__((ext_vector_type(16)));

static __device__ __forceinline__ v8f wmma_bf16(v16bf a, v16bf b, v8f c) {
  // (neg_a, A, neg_b, B, c_mod, C, reuse_a, reuse_b)
  return __builtin_amdgcn_wmma_f32_16x16x32_bf16(false, a, false, b, (short)0,
                                                 c, false, false);
}
static __device__ __forceinline__ v8f wmma_f32(v2f a, v2f b, v8f c) {
  return __builtin_amdgcn_wmma_f32_16x16x4_f32(false, a, false, b, (short)0, c,
                                               false, false);
}

// ---------------------------------------------------------------------------
// Prep: pack weights into exact A-operand layout, split into bf16 hi/lo.
// wsA[tap(kh*4+kw)][co_tile][p(hi/lo)][lane][e] ; e->K map: K=16*h+e
// Total 16*2*2*32*16 bf16 = 64 KB.
// ---------------------------------------------------------------------------
__global__ __launch_bounds__(256) void pack_weights_kernel(
    const float* __restrict__ w, __bf16* __restrict__ wsA) {
  const int idx = blockIdx.x * 256 + threadIdx.x;  // [tap][t][lane][e] = 16384
  if (idx >= 16 * 2 * 32 * 16) return;
  const int e    = idx & 15;
  const int lane = (idx >> 4) & 31;
  const int t    = (idx >> 9) & 1;
  const int tap  = idx >> 10;
  const int kh = tap >> 2, kw = tap & 3;
  const int h = lane >> 4, m = lane & 15;
  const int K = 16 * h + e;  // ci index (A 16x32 layout)
  const float val = w[(size_t)K * 512 + (t * 16 + m) * 16 + kh * 4 + kw];
  const __bf16 hi = (__bf16)val;
  const __bf16 lo = (__bf16)(val - (float)hi);
  wsA[(((size_t)(tap * 2 + t) * 2 + 0) * 32 + lane) * 16 + e] = hi;
  wsA[(((size_t)(tap * 2 + t) * 2 + 1) * 32 + lane) * 16 + e] = lo;
}

// ---------------------------------------------------------------------------
// Main: conv_transpose2d via bf16x3-split WMMA (hi*hi + hi*lo + lo*hi, f32 acc)
// One block = one (b, oh', r) output row: stages 2 input rows (with zero halo)
// into LDS in B-operand layout, then 8 waves each compute 32(Cout)x32(ow).
// ---------------------------------------------------------------------------
__global__ __launch_bounds__(256, 1) void convt2d_bf16x3_kernel(
    const float* __restrict__ x, const float* __restrict__ bias,
    float* __restrict__ y, const __bf16* __restrict__ wsA) {
  constexpr int Cin = 32, Cout = 32, H = 128, W = 128, OH = 256, OW = 256;
  // LDS layout: [th][iwi 0..129][p][ci], ci contiguous (B-operand order)
  __shared__ alignas(64) __bf16 smem[2 * 130 * 2 * 32];  // 33,280 B

  const int bid = blockIdx.x;        // 4096 = 16 b * 128 oh' * 2 r
  const int r   = bid & 1;
  const int ohp = (bid >> 1) & 127;
  const int b   = bid >> 8;

  const int ih0 = ohp;                   const int kh0 = r ? 2 : 1;
  const int ih1 = r ? ohp + 1 : ohp - 1; const int kh1 = r ? 0 : 3;

  // ---- stage x rows (pre-split bf16 hi/lo, zero-filled halo & OOB rows) ----
  for (int idx = threadIdx.x; idx < 2 * Cin * 130; idx += 256) {
    const int th  = idx / (Cin * 130);
    const int rem = idx - th * Cin * 130;
    const int ci  = rem / 130;
    const int iwi = rem - ci * 130;
    const int iw  = iwi - 1;
    const int ih  = th ? ih1 : ih0;
    float val = 0.0f;
    if (ih >= 0 && ih < H && iw >= 0 && iw < W)
      val = x[((size_t)(b * Cin + ci) * H + ih) * W + iw];
    const __bf16 hi = (__bf16)val;
    const __bf16 lo = (__bf16)(val - (float)hi);
    const int base = ((th * 130 + iwi) * 2) * 32 + ci;
    smem[base]      = hi;
    smem[base + 32] = lo;
  }
  __syncthreads();

  // ---- per-wave WMMA tile: 32 Cout x 16 ow' x 2 s-parities ----
  const int lane = threadIdx.x & 31;
  const int h    = lane >> 4;
  const int i16  = lane & 15;
  const int tile = threadIdx.x >> 5;  // 8 waves = 8 tiles of 16 ow'
  const int owb  = tile * 16;

  v8f acc[2][2] = {};  // [co_tile][s]

#pragma unroll
  for (int s = 0; s < 2; ++s) {
#pragma unroll
    for (int tw = 0; tw < 2; ++tw) {
      const int kw  = s ? (tw ? 0 : 2) : (tw ? 3 : 1);
      const int dw  = s ? (tw ? 1 : 0) : (tw ? -1 : 0);
      const int iwi = owb + i16 + dw + 1;  // always in [0,130)
#pragma unroll
      for (int th = 0; th < 2; ++th) {
        const int kh  = th ? kh1 : kh0;
        const int tap = kh * 4 + kw;
        // B operands: K = ci = 16*h + e, contiguous 32B in LDS
        const __bf16* bp = &smem[((th * 130 + iwi) * 2) * 32 + h * 16];
        const v16bf Bh = *reinterpret_cast<const v16bf*>(bp);
        const v16bf Bl = *reinterpret_cast<const v16bf*>(bp + 64);
#pragma unroll
        for (int t = 0; t < 2; ++t) {
          const __bf16* ap =
              wsA + (((size_t)(tap * 2 + t) * 2) * 32 + lane) * 16;
          const v16bf Ah = *reinterpret_cast<const v16bf*>(ap);
          const v16bf Al = *reinterpret_cast<const v16bf*>(ap + 32 * 16);
          acc[t][s] = wmma_bf16(Ah, Bh, acc[t][s]);  // hi*hi
          acc[t][s] = wmma_bf16(Ah, Bl, acc[t][s]);  // hi*lo
          acc[t][s] = wmma_bf16(Al, Bh, acc[t][s]);  // lo*hi
        }
      }
    }
  }

  // ---- epilogue: bias + packed (s0,s1) float2 stores, fully coalesced ----
  const int oh  = 2 * ohp + r;
  const int ow0 = 2 * (owb + i16);
#pragma unroll
  for (int t = 0; t < 2; ++t) {
#pragma unroll
    for (int v = 0; v < 8; ++v) {
      const int co = t * 16 + 8 * h + v;  // C/D layout: M = v + 8*half
      const float bv = bias[co];
      float2 o;
      o.x = acc[t][0][v] + bv;
      o.y = acc[t][1][v] + bv;
      float* yp = y + (((size_t)(b * Cout + co) * OH) + oh) * OW + ow0;
      *reinterpret_cast<float2*>(yp) = o;
    }
  }
}

// ---------------------------------------------------------------------------
// Fallback (no workspace): round-1 fp32 WMMA kernel, full precision.
// ---------------------------------------------------------------------------
__global__ __launch_bounds__(256, 1) void convt2d_f32wmma_kernel(
    const float* __restrict__ x, const float* __restrict__ w,
    const float* __restrict__ bias, float* __restrict__ y) {
  constexpr int Cin = 32, Cout = 32, H = 128, W = 128, OH = 256, OW = 256;
  constexpr int HW = H * W;
  const int lane = threadIdx.x & 31;
  const int half = lane >> 4;
  const int i16  = lane & 15;
  const int wid  = blockIdx.x * (blockDim.x >> 5) + (threadIdx.x >> 5);
  const int tile = wid & 7;
  const int r    = (wid >> 3) & 1;
  const int ohp  = (wid >> 4) & 127;
  const int b    = wid >> 11;
  const int owb = tile * 16;
  const int oh  = 2 * ohp + r;
  const int kA  = half * 2;
  const int ih0 = ohp;                   const int kh0 = r ? 2 : 1;
  const int ih1 = r ? ohp + 1 : ohp - 1; const int kh1 = r ? 0 : 3;
  const bool ih1ok = (ih1 >= 0) && (ih1 < H);
  v8f acc[2][2] = {};
#pragma unroll
  for (int s = 0; s < 2; ++s) {
#pragma unroll
    for (int tw = 0; tw < 2; ++tw) {
      const int kw = s ? (tw ? 0 : 2) : (tw ? 3 : 1);
      const int dw = s ? (tw ? 1 : 0) : (tw ? -1 : 0);
      const int iw = owb + i16 + dw;
      const float msk = (iw >= 0 && iw < W) ? 1.0f : 0.0f;
      const int iwc = iw < 0 ? 0 : (iw > W - 1 ? W - 1 : iw);
#pragma unroll
      for (int th = 0; th < 2; ++th) {
        if (th && !ih1ok) continue;
        const int ih = th ? ih1 : ih0;
        const int kh = th ? kh1 : kh0;
        const float* xp0 = x + ((size_t)(b * Cin + kA) * H + ih) * W + iwc;
        const float* wp0 = w + (size_t)kA * 512 + i16 * 16 + kh * 4 + kw;
#pragma unroll
        for (int c = 0; c < 8; ++c) {
          const float* xp = xp0 + (size_t)(4 * c) * HW;
          v2f B; B.x = msk * xp[0]; B.y = msk * xp[HW];
          const float* wp = wp0 + (size_t)(4 * c) * 512;
          v2f A0; A0.x = wp[0];   A0.y = wp[512];
          v2f A1; A1.x = wp[256]; A1.y = wp[256 + 512];
          acc[0][s] = wmma_f32(A0, B, acc[0][s]);
          acc[1][s] = wmma_f32(A1, B, acc[1][s]);
        }
      }
    }
  }
  const int ow0 = 2 * (owb + i16);
#pragma unroll
  for (int t = 0; t < 2; ++t) {
#pragma unroll
    for (int v = 0; v < 8; ++v) {
      const int co = t * 16 + 8 * half + v;
      const float bv = bias[co];
      float2 o;
      o.x = acc[t][0][v] + bv;
      o.y = acc[t][1][v] + bv;
      float* yp = y + (((size_t)(b * Cout + co) * OH) + oh) * OW + ow0;
      *reinterpret_cast<float2*>(yp) = o;
    }
  }
}

extern "C" void kernel_launch(void* const* d_in, const int* in_sizes, int n_in,
                              void* d_out, int out_size, void* d_ws,
                              size_t ws_size, hipStream_t stream) {
  const float* x    = (const float*)d_in[0];  // (16,32,128,128)
  const float* w    = (const float*)d_in[1];  // (32,32,4,4)
  const float* bias = (const float*)d_in[2];  // (32)
  float* y = (float*)d_out;                   // (16,32,256,256)

  constexpr size_t WSA_BYTES = 16 * 2 * 2 * 32 * 16 * sizeof(__bf16);  // 64 KB
  if (ws_size >= WSA_BYTES) {
    __bf16* wsA = (__bf16*)d_ws;
    pack_weights_kernel<<<dim3(64), dim3(256), 0, stream>>>(w, wsA);
    convt2d_bf16x3_kernel<<<dim3(4096), dim3(256), 0, stream>>>(x, bias, y,
                                                                wsA);
  } else {
    convt2d_f32wmma_kernel<<<dim3(4096), dim3(256), 0, stream>>>(x, w, bias,
                                                                 y);
  }
}
